// DecompositionBlock_13649406067517
// MI455X (gfx1250) — compile-verified, matrix operand order checked by reading, not run on hardware
//
#include <hip/hip_runtime.h>
#include <hip/hip_bf16.h>

typedef __attribute__((ext_vector_type(16))) _Float16 v16h;
typedef __attribute__((ext_vector_type(4)))  _Float16 v4h;
typedef __attribute__((ext_vector_type(8)))  float    v8f;

#define BATCH 32
#define SEQ   1024
#define DMODEL 512
#define NF    8
#define NKNOT 32
#define NHEAD 8
#define HD    64
#define MROWS (BATCH*SEQ)          /* 32768 */
#define QKVN  (3*DMODEL)           /* 1536  */

static __device__ __forceinline__ int wmma_kidx(int j, int hs) {
  return ((j < 4) ? (2 * j) : (8 + 2 * j)) + hs * 8;
}

static __device__ __forceinline__ unsigned encodeF(float f) {
  unsigned u = __float_as_uint(f);
  return (u & 0x80000000u) ? ~u : (u | 0x80000000u);
}
static __device__ __forceinline__ float decodeF(unsigned u) {
  unsigned b = (u & 0x80000000u) ? (u ^ 0x80000000u) : ~u;
  return __uint_as_float(b);
}

// CDNA5 async global->LDS copy (ASYNCcnt-tracked, §15.18.3).
// LDS operand = low 32 bits of the flat shared-space address (aperture mapping).
static __device__ __forceinline__ void async_cp_b128(void* lds, const void* g) {
  unsigned l = (unsigned)(unsigned long long)lds;
  asm volatile("global_load_async_to_lds_b128 %0, %1, off" :: "v"(l), "v"(g) : "memory");
}
static __device__ __forceinline__ void wait_async0() {
#if __has_builtin(__builtin_amdgcn_s_wait_asynccnt)
  __builtin_amdgcn_s_wait_asynccnt(0);
#else
  asm volatile("s_wait_asynccnt 0x0" ::: "memory");
#endif
}

// A-fragment (16x32 f16): row fixed per lane, K contiguous from `base`
static __device__ __forceinline__ v16h load_frag_row(const _Float16* base, int hs) {
  v16h a;
#pragma unroll
  for (int j = 0; j < 8; ++j) {
    int kk = wmma_kidx(j, hs);
    a[2 * j]     = base[kk];
    a[2 * j + 1] = base[kk + 1];
  }
  return a;
}
// B-fragment (32x16 f16): column fixed per lane, K strided by ld from `base`
static __device__ __forceinline__ v16h load_frag_col(const _Float16* base, long ld, int hs) {
  v16h b;
#pragma unroll
  for (int j = 0; j < 8; ++j) {
    int kk = wmma_kidx(j, hs);
    b[2 * j]     = base[(long)kk * ld];
    b[2 * j + 1] = base[(long)(kk + 1) * ld];
  }
  return b;
}

// ---------------------------------------------------------------- casts
__global__ __launch_bounds__(256) void k_cast_x(const float* __restrict__ x,
                                                _Float16* __restrict__ xh, int n4) {
  int i = blockIdx.x * 256 + threadIdx.x;
  if (i < n4) {
    float4 v = ((const float4*)x)[i];
    v4h h;
    h[0] = (_Float16)v.x; h[1] = (_Float16)v.y;
    h[2] = (_Float16)v.z; h[3] = (_Float16)v.w;
    ((v4h*)xh)[i] = h;
  }
}

// dst[k*Ndim + n] = (f16) src[n*Kdim + k]   (src is (Ndim x Kdim) row-major)
__global__ __launch_bounds__(256) void k_tcast(const float* __restrict__ src,
                                               _Float16* __restrict__ dst,
                                               int Kdim, int Ndim) {
  int i = blockIdx.x * 256 + threadIdx.x;
  if (i < Kdim * Ndim) {
    int k = i / Ndim, n = i - k * Ndim;
    dst[i] = (_Float16)src[(size_t)n * Kdim + k];
  }
}

// ---------------------------------------------------------------- WMMA GEMM
// C(M,N) = A(M,K) x B(K,N) + bias[n];  A,B f16 row-major.
// 128x128x32 tiles, double-buffered LDS fed by async global->LDS copies.
template <bool OUT_F16>
__global__ __launch_bounds__(256)
void k_gemm(const _Float16* __restrict__ A, const _Float16* __restrict__ Bm,
            const float* __restrict__ bias, float* __restrict__ Cf,
            _Float16* __restrict__ Ch, int M, int N, int K) {
  constexpr int BM = 128, BN = 128, BK = 32;
  constexpr int LDA = BK + 8;   // 40 halfs (80B, 16B aligned, conflict-free)
  constexpr int LDB = BN + 8;   // 136 halfs (272B)
  __shared__ _Float16 As[2][BM * LDA];
  __shared__ _Float16 Bs[2][BK * LDB];

  const int tid = threadIdx.x;
  const int wave = tid >> 5, lane = tid & 31;
  const int wm = (wave >> 2) * 64;  // 0 / 64
  const int wn = (wave & 3) * 32;   // 0 / 32 / 64 / 96
  const int r16 = lane & 15, hs = lane >> 4;
  const int bm0 = blockIdx.y * BM, bn0 = blockIdx.x * BN;

  // per-thread copy slots (each 8 halfs = 16B)
  const int arow0 = tid >> 2,        ac8 = (tid & 3) * 8;          // A: rows tid/4, tid/4+64
  const int brow0 = tid >> 4,        bc8 = (tid & 15) * 8;         // B: rows tid/16, tid/16+16

  v8f zero = {};
  v8f c[4][2];
#pragma unroll
  for (int i = 0; i < 4; ++i) { c[i][0] = zero; c[i][1] = zero; }

  const int KT = K / BK;
  // prime buffer 0
  {
    async_cp_b128(&As[0][arow0 * LDA + ac8],        A + (size_t)(bm0 + arow0) * K + ac8);
    async_cp_b128(&As[0][(arow0 + 64) * LDA + ac8], A + (size_t)(bm0 + arow0 + 64) * K + ac8);
    async_cp_b128(&Bs[0][brow0 * LDB + bc8],        Bm + (size_t)brow0 * N + bn0 + bc8);
    async_cp_b128(&Bs[0][(brow0 + 16) * LDB + bc8], Bm + (size_t)(brow0 + 16) * N + bn0 + bc8);
  }

  for (int kt = 0; kt < KT; ++kt) {
    wait_async0();        // my async copies into buf[kt&1] are done
    __syncthreads();      // ... and everyone else's too; also: all reads of buf[(kt+1)&1] finished
    const int cur = kt & 1, nxt = cur ^ 1;
    if (kt + 1 < KT) {    // overlap next tile's copies with this tile's WMMAs
      int k0n = (kt + 1) * BK;
      async_cp_b128(&As[nxt][arow0 * LDA + ac8],        A + (size_t)(bm0 + arow0) * K + k0n + ac8);
      async_cp_b128(&As[nxt][(arow0 + 64) * LDA + ac8], A + (size_t)(bm0 + arow0 + 64) * K + k0n + ac8);
      async_cp_b128(&Bs[nxt][brow0 * LDB + bc8],        Bm + (size_t)(k0n + brow0) * N + bn0 + bc8);
      async_cp_b128(&Bs[nxt][(brow0 + 16) * LDB + bc8], Bm + (size_t)(k0n + brow0 + 16) * N + bn0 + bc8);
    }
    v16h bf0 = load_frag_col(&Bs[cur][wn + 0  + r16], LDB, hs);
    v16h bf1 = load_frag_col(&Bs[cur][wn + 16 + r16], LDB, hs);
#pragma unroll
    for (int mt = 0; mt < 4; ++mt) {
      v16h af = load_frag_row(&As[cur][(wm + mt * 16 + r16) * LDA], hs);
      c[mt][0] = __builtin_amdgcn_wmma_f32_16x16x32_f16(false, af, false, bf0,
                                                        (short)0, c[mt][0], false, false);
      c[mt][1] = __builtin_amdgcn_wmma_f32_16x16x32_f16(false, af, false, bf1,
                                                        (short)0, c[mt][1], false, false);
    }
  }

#pragma unroll
  for (int mt = 0; mt < 4; ++mt)
#pragma unroll
    for (int nt = 0; nt < 2; ++nt) {
      int ncol = bn0 + wn + nt * 16 + r16;
      float bv = bias[ncol];
#pragma unroll
      for (int e = 0; e < 8; ++e) {
        int mrow = bm0 + wm + mt * 16 + e + hs * 8;
        float val = c[mt][nt][e] + bv;
        if (OUT_F16) Ch[(size_t)mrow * N + ncol] = (_Float16)val;
        else         Cf[(size_t)mrow * N + ncol] = val;
      }
    }
}

// ---------------------------------------------------------------- attention
// qkv: (32768,1536) f16, rows = l*SEQ+n.  One wave per (n,h).
__global__ __launch_bounds__(128)
void k_attn(const _Float16* __restrict__ qkv, _Float16* __restrict__ aout) {
  __shared__ float    sc[4][32 * 33];
  __shared__ _Float16 atb[4][32 * 34];

  const int wave = threadIdx.x >> 5, lane = threadIdx.x & 31;
  const int wid = blockIdx.x * 4 + wave;
  const int n = wid >> 3, h = wid & 7;
  const size_t lstride = (size_t)SEQ * QKVN;
  const _Float16* qp = qkv + (size_t)n * QKVN + h * HD;
  const _Float16* kp = qp + DMODEL;
  const _Float16* vp = qp + 2 * DMODEL;
  const int r16 = lane & 15, hs = lane >> 4;
  float* s = sc[wave];
  _Float16* a = atb[wave];
  v8f zero = {};

  // scores = (q*0.125) k^T : M=32(l) N=32(m') K=64(d)
#pragma unroll
  for (int mt = 0; mt < 2; ++mt)
#pragma unroll
    for (int nt = 0; nt < 2; ++nt) {
      v8f acc = zero;
#pragma unroll
      for (int d0 = 0; d0 < 64; d0 += 32) {
        v16h af = load_frag_row(qp + (size_t)(mt * 16 + r16) * lstride + d0, hs);
        v16h bf = load_frag_row(kp + (size_t)(nt * 16 + r16) * lstride + d0, hs);
        acc = __builtin_amdgcn_wmma_f32_16x16x32_f16(false, af, false, bf,
                                                     (short)0, acc, false, false);
      }
#pragma unroll
      for (int e = 0; e < 8; ++e)
        s[(mt * 16 + e + hs * 8) * 33 + nt * 16 + r16] = acc[e] * 0.125f;
    }
  __syncthreads();

  // softmax: lane owns row `lane`
  {
    float* row = s + lane * 33;
    float mx = row[0];
#pragma unroll
    for (int i = 1; i < 32; ++i) mx = fmaxf(mx, row[i]);
    float sum = 0.f;
#pragma unroll
    for (int i = 0; i < 32; ++i) sum += __expf(row[i] - mx);
    float inv = 1.f / sum;
    _Float16* arow = a + lane * 34;
#pragma unroll
    for (int i = 0; i < 32; ++i) arow[i] = (_Float16)(__expf(row[i] - mx) * inv);
  }
  __syncthreads();

  // out = attn @ v : M=32(l) N=64(d) K=32(m')
#pragma unroll
  for (int dt = 0; dt < 4; ++dt) {
    v16h bf = load_frag_col(vp + dt * 16 + r16, (long)lstride, hs);
#pragma unroll
    for (int mt = 0; mt < 2; ++mt) {
      v16h af = load_frag_row(a + (mt * 16 + r16) * 34, hs);
      v8f acc = zero;
      acc = __builtin_amdgcn_wmma_f32_16x16x32_f16(false, af, false, bf,
                                                   (short)0, acc, false, false);
#pragma unroll
      for (int e = 0; e < 8; ++e) {
        int l = mt * 16 + e + hs * 8;
        int d = dt * 16 + r16;
        aout[((size_t)l * SEQ + n) * DMODEL + h * HD + d] = (_Float16)acc[e];
      }
    }
  }
}

// ---------------------------------------------------------------- BN channel stats (C=512)
__global__ __launch_bounds__(256)
void k_cstats(const float* __restrict__ X, float* __restrict__ sum, float* __restrict__ sq) {
  const int r0 = blockIdx.x * 128;
  const int c0 = threadIdx.x, c1 = threadIdx.x + 256;
  float s0 = 0, q0 = 0, s1 = 0, q1 = 0;
  for (int r = 0; r < 128; ++r) {
    const float* row = X + (size_t)(r0 + r) * DMODEL;
    float av = row[c0]; s0 += av; q0 += av * av;
    float bv = row[c1]; s1 += bv; q1 += bv * bv;
  }
  atomicAdd(&sum[c0], s0); atomicAdd(&sq[c0], q0);
  atomicAdd(&sum[c1], s1); atomicAdd(&sq[c1], q1);
}

// ---------------------------------------------------------------- projected = x @ proj_w.T + b
__global__ __launch_bounds__(64)
void k_proj(const float* __restrict__ x, const float* __restrict__ pw,
            const float* __restrict__ pb, float* __restrict__ p) {
  __shared__ float red[64 * NF];
  const int r = blockIdx.x;
  const float* xr = x + (size_t)r * DMODEL;
  float acc[NF];
#pragma unroll
  for (int f = 0; f < NF; ++f) acc[f] = 0.f;
  for (int cidx = threadIdx.x; cidx < DMODEL; cidx += 64) {
    float xv = xr[cidx];
#pragma unroll
    for (int f = 0; f < NF; ++f) acc[f] += xv * pw[f * DMODEL + cidx];
  }
#pragma unroll
  for (int f = 0; f < NF; ++f) red[threadIdx.x * NF + f] = acc[f];
  __syncthreads();
  if (threadIdx.x < NF) {
    float s = 0.f;
    for (int t = 0; t < 64; ++t) s += red[t * NF + threadIdx.x];
    p[(size_t)r * NF + threadIdx.x] = s + pb[threadIdx.x];
  }
}

// per-feature sum/sq/min/max of projected
__global__ __launch_bounds__(256)
void k_pstats(const float* __restrict__ p, float* __restrict__ psum, float* __restrict__ psq,
              unsigned* __restrict__ pmin, unsigned* __restrict__ pmax) {
  int r = blockIdx.x * 256 + threadIdx.x;
  const float* pr = p + (size_t)r * NF;
#pragma unroll
  for (int f = 0; f < NF; ++f) {
    float v = pr[f];
    atomicAdd(&psum[f], v);
    atomicAdd(&psq[f], v * v);
    atomicMin(&pmin[f], encodeF(v));
    atomicMax(&pmax[f], encodeF(v));
  }
}

// G = W^T W (8x8), wbv = W^T b, colw = colsum(W), gbsc = {sum b^2, sum b}
__global__ __launch_bounds__(256)
void k_gatepre(const float* __restrict__ gw, const float* __restrict__ gb,
               float* __restrict__ G, float* __restrict__ wbv,
               float* __restrict__ colw, float* __restrict__ gbsc) {
  float Gl[64], wl[NF], cl[NF];
#pragma unroll
  for (int i = 0; i < 64; ++i) Gl[i] = 0.f;
#pragma unroll
  for (int f = 0; f < NF; ++f) { wl[f] = 0.f; cl[f] = 0.f; }
  float b2 = 0.f, bs = 0.f;
  for (int j = threadIdx.x; j < 2 * DMODEL; j += 256) {
    float w[NF];
#pragma unroll
    for (int f = 0; f < NF; ++f) w[f] = gw[(size_t)j * NF + f];
    float bj = gb[j];
#pragma unroll
    for (int f = 0; f < NF; ++f) {
      cl[f] += w[f];
      wl[f] += w[f] * bj;
#pragma unroll
      for (int g = 0; g < NF; ++g) Gl[f * NF + g] += w[f] * w[g];
    }
    b2 += bj * bj; bs += bj;
  }
#pragma unroll
  for (int i = 0; i < 64; ++i) atomicAdd(&G[i], Gl[i]);
#pragma unroll
  for (int f = 0; f < NF; ++f) { atomicAdd(&wbv[f], wl[f]); atomicAdd(&colw[f], cl[f]); }
  atomicAdd(&gbsc[0], b2); atomicAdd(&gbsc[1], bs);
}

// BN(projected) + spline -> t
__global__ __launch_bounds__(256)
void k_spline(const float* __restrict__ p, const float* __restrict__ psum,
              const float* __restrict__ psq, const unsigned* __restrict__ pmin,
              const unsigned* __restrict__ pmax, const float* __restrict__ png,
              const float* __restrict__ pnb, const float* __restrict__ knots,
              const float* __restrict__ cps, float* __restrict__ tout) {
  int i = blockIdx.x * 256 + threadIdx.x;   // over MROWS*NF
  int f = i & (NF - 1);
  const float cnt = (float)MROWS;
  float mean = psum[f] / cnt;
  float var = psq[f] / cnt - mean * mean;
  float inv = rsqrtf(var + 1e-5f) * png[f];
  float beta = pnb[f];
  float v = (p[i] - mean) * inv + beta;
  float c1 = (decodeF(pmin[f]) - mean) * inv + beta;
  float c2 = (decodeF(pmax[f]) - mean) * inv + beta;
  float lo = fminf(c1, c2), hi = fmaxf(c1, c2);
  float xn = (v - lo) / (hi - lo + 1e-6f);
  const float* kn = knots + f * NKNOT;
  const float* cp = cps + f * NKNOT;
  int idx = 0;
#pragma unroll
  for (int k = 1; k < NKNOT; ++k) idx += (xn >= kn[k]) ? 1 : 0;
  idx = min(idx, NKNOT - 2);
  float k0 = kn[idx], k1 = kn[idx + 1];
  float tt = (xn - k0) / (k1 - k0);
  float val = (1.f - tt) * cp[idx] + tt * cp[idx + 1];
  bool inr = (xn >= kn[0]) && (xn <= kn[NKNOT - 1]);
  tout[i] = inr ? val : 0.f;
}

// analytic per-s stats of h = t W^T + b over (b, j)
__global__ __launch_bounds__(256)
void k_gbnstats(const float* __restrict__ t, const float* __restrict__ G,
                const float* __restrict__ wbv, const float* __restrict__ colw,
                const float* __restrict__ gbsc, float* __restrict__ ssum,
                float* __restrict__ ssq) {
  int s = blockIdx.x * 256 + threadIdx.x;
  if (s >= SEQ) return;
  float Gl[64], wl[NF], cw[NF];
#pragma unroll
  for (int i = 0; i < 64; ++i) Gl[i] = G[i];
#pragma unroll
  for (int f = 0; f < NF; ++f) { wl[f] = wbv[f]; cw[f] = colw[f]; }
  float gb2 = gbsc[0], gbs = gbsc[1];
  float lin = 0.f, quad = 0.f;
  for (int b = 0; b < BATCH; ++b) {
    const float* tv = t + ((size_t)b * SEQ + s) * NF;
    float tf[NF];
#pragma unroll
    for (int f = 0; f < NF; ++f) tf[f] = tv[f];
    float q = 0.f;
#pragma unroll
    for (int f = 0; f < NF; ++f) {
      lin += tf[f] * cw[f];
      q += 2.f * tf[f] * wl[f];
#pragma unroll
      for (int g = 0; g < NF; ++g) q += tf[f] * Gl[f * NF + g] * tf[g];
    }
    quad += q + gb2;
  }
  ssum[s] = lin + (float)BATCH * gbs;
  ssq[s] = quad;
}

// fused: h -> seq-BN -> GLU -> LayerNorm -> f16
__global__ __launch_bounds__(256)
void k_gluln(const float* __restrict__ t, const float* __restrict__ gw,
             const float* __restrict__ gb, const float* __restrict__ ssum,
             const float* __restrict__ ssq, const float* __restrict__ gbng,
             const float* __restrict__ gbnb, const float* __restrict__ lng,
             const float* __restrict__ lnb, _Float16* __restrict__ gated) {
  __shared__ float tf[NF];
  __shared__ float rs[8], rq[8];
  const int r = blockIdx.x;
  const int s = r & (SEQ - 1);
  if (threadIdx.x < NF) tf[threadIdx.x] = t[(size_t)r * NF + threadIdx.x];
  __syncthreads();
  const float cnt = (float)(BATCH * 2 * DMODEL);
  float mean = ssum[s] / cnt;
  float var = ssq[s] / cnt - mean * mean;
  float inv = rsqrtf(var + 1e-5f);
  float g = gbng[s], be = gbnb[s];

  float glu[2];
#pragma unroll
  for (int i = 0; i < 2; ++i) {
    int j = threadIdx.x + i * 256;
    float av = gb[j], bv = gb[j + DMODEL];
#pragma unroll
    for (int f = 0; f < NF; ++f) {
      av += tf[f] * gw[(size_t)j * NF + f];
      bv += tf[f] * gw[(size_t)(j + DMODEL) * NF + f];
    }
    av = (av - mean) * inv * g + be;
    bv = (bv - mean) * inv * g + be;
    glu[i] = av * (1.f / (1.f + __expf(-bv)));
  }
  float sum = glu[0] + glu[1];
  float sq = glu[0] * glu[0] + glu[1] * glu[1];
#pragma unroll
  for (int m = 16; m >= 1; m >>= 1) {
    sum += __shfl_xor(sum, m, 32);
    sq += __shfl_xor(sq, m, 32);
  }
  const int wave = threadIdx.x >> 5, lane = threadIdx.x & 31;
  if (lane == 0) { rs[wave] = sum; rq[wave] = sq; }
  __syncthreads();
  float tot = 0.f, totq = 0.f;
#pragma unroll
  for (int w = 0; w < 8; ++w) { tot += rs[w]; totq += rq[w]; }
  float m = tot * (1.f / DMODEL);
  float v = totq * (1.f / DMODEL) - m * m;
  float inv2 = rsqrtf(v + 1e-5f);
#pragma unroll
  for (int i = 0; i < 2; ++i) {
    int j = threadIdx.x + i * 256;
    float val = (glu[i] - m) * inv2 * lng[j] + lnb[j];
    gated[(size_t)r * DMODEL + j] = (_Float16)val;
  }
}

// out = x + BN(attn_out) + BN(outbuf)
__global__ __launch_bounds__(256)
void k_final(const float* __restrict__ x, const float* __restrict__ ao,
             const float* __restrict__ ob, const float* __restrict__ stats,
             const float* __restrict__ ag, const float* __restrict__ abta,
             const float* __restrict__ og, const float* __restrict__ obta,
             float* __restrict__ out) {
  size_t i = (size_t)blockIdx.x * 256 + threadIdx.x;
  int c = (int)(i & (DMODEL - 1));
  const float ic = 1.f / (float)MROWS;
  float m1 = stats[c] * ic;
  float v1 = stats[512 + c] * ic - m1 * m1;
  float m2 = stats[1024 + c] * ic;
  float v2 = stats[1536 + c] * ic - m2 * m2;
  float y1 = (ao[i] - m1) * rsqrtf(v1 + 1e-5f) * ag[c] + abta[c];
  float y2 = (ob[i] - m2) * rsqrtf(v2 + 1e-5f) * og[c] + obta[c];
  out[i] = x[i] + y1 + y2;
}

#define STATS_FLOATS 4224
__global__ __launch_bounds__(256) void k_init(float* __restrict__ stats) {
  int i = blockIdx.x * 256 + threadIdx.x;
  if (i < STATS_FLOATS) stats[i] = 0.f;
  if (i >= 2064 && i < 2072) ((unsigned*)stats)[i] = 0xFFFFFFFFu;  // pmin encoded init
}

// ---------------------------------------------------------------- launch
extern "C" void kernel_launch(void* const* d_in, const int* in_sizes, int n_in,
                              void* d_out, int out_size, void* d_ws, size_t ws_size,
                              hipStream_t stream) {
  const float* x      = (const float*)d_in[0];
  const float* in_w   = (const float*)d_in[1];
  const float* in_b   = (const float*)d_in[2];
  const float* out_w  = (const float*)d_in[3];
  const float* out_b  = (const float*)d_in[4];
  const float* attn_g = (const float*)d_in[5];
  const float* attn_b = (const float*)d_in[6];
  const float* proj_w = (const float*)d_in[7];
  const float* proj_b = (const float*)d_in[8];
  const float* pn_g   = (const float*)d_in[9];
  const float* pn_b   = (const float*)d_in[10];
  const float* knots  = (const float*)d_in[11];
  const float* cps    = (const float*)d_in[12];
  const float* gate_w = (const float*)d_in[13];
  const float* gate_b = (const float*)d_in[14];
  const float* gbn_g  = (const float*)d_in[15];
  const float* gbn_b  = (const float*)d_in[16];
  const float* ln_g   = (const float*)d_in[17];
  const float* ln_b   = (const float*)d_in[18];
  const float* outp_w = (const float*)d_in[19];
  const float* outp_b = (const float*)d_in[20];
  const float* on_g   = (const float*)d_in[21];
  const float* on_b   = (const float*)d_in[22];
  float* out = (float*)d_out;
  (void)in_sizes; (void)n_in; (void)out_size; (void)ws_size;

  char* base = (char*)d_ws;
  size_t off = 0;
  auto carve = [&](size_t bytes) { char* p = base + off; off = (off + bytes + 255) & ~(size_t)255; return p; };
  _Float16* xh     = (_Float16*)carve((size_t)MROWS * DMODEL * 2);
  _Float16* wqkvT  = (_Float16*)carve((size_t)DMODEL * QKVN * 2);
  _Float16* woutT  = (_Float16*)carve((size_t)DMODEL * DMODEL * 2);
  _Float16* woutpT = (_Float16*)carve((size_t)DMODEL * DMODEL * 2);
  _Float16* qkvh   = (_Float16*)carve((size_t)MROWS * QKVN * 2);
  _Float16* attnh  = (_Float16*)carve((size_t)MROWS * DMODEL * 2);
  float*    attn_o = (float*)   carve((size_t)MROWS * DMODEL * 4);
  float*    pbuf   = (float*)   carve((size_t)MROWS * NF * 4);
  float*    tbuf   = (float*)   carve((size_t)MROWS * NF * 4);
  _Float16* gatedh = (_Float16*)carve((size_t)MROWS * DMODEL * 2);
  float*    outbuf = (float*)   carve((size_t)MROWS * DMODEL * 4);
  float*    stats  = (float*)   carve((size_t)STATS_FLOATS * 4);

  float* s1 = stats;            float* q1 = stats + 512;
  float* s2 = stats + 1024;     float* q2 = stats + 1536;
  float* psum = stats + 2048;   float* psq = stats + 2056;
  unsigned* pminE = (unsigned*)(stats + 2064);
  unsigned* pmaxE = (unsigned*)(stats + 2072);
  float* G = stats + 2080;      float* wbv = stats + 2144;
  float* colw = stats + 2152;   float* gbsc = stats + 2160;
  float* ssum = stats + 2176;   float* ssq = stats + 3200;

  k_init<<<(STATS_FLOATS + 255) / 256, 256, 0, stream>>>(stats);
  k_cast_x<<<(MROWS * DMODEL / 4 + 255) / 256, 256, 0, stream>>>(x, xh, MROWS * DMODEL / 4);
  k_tcast<<<(DMODEL * QKVN + 255) / 256, 256, 0, stream>>>(in_w, wqkvT, DMODEL, QKVN);
  k_tcast<<<(DMODEL * DMODEL + 255) / 256, 256, 0, stream>>>(out_w, woutT, DMODEL, DMODEL);
  k_tcast<<<(DMODEL * DMODEL + 255) / 256, 256, 0, stream>>>(outp_w, woutpT, DMODEL, DMODEL);

  // qkv = x @ in_w.T + in_b   (f16 out)
  k_gemm<true><<<dim3(QKVN / 128, MROWS / 128), 256, 0, stream>>>(
      xh, wqkvT, in_b, nullptr, qkvh, MROWS, QKVN, DMODEL);

  // attention (one wave per (n,h))
  k_attn<<<(SEQ * NHEAD) / 4, 128, 0, stream>>>(qkvh, attnh);

  // attn_out = attnh @ out_w.T + out_b (f32 out) + channel stats
  k_gemm<false><<<dim3(DMODEL / 128, MROWS / 128), 256, 0, stream>>>(
      attnh, woutT, out_b, attn_o, nullptr, MROWS, DMODEL, DMODEL);
  k_cstats<<<MROWS / 128, 256, 0, stream>>>(attn_o, s1, q1);

  // projection branch
  k_proj<<<MROWS, 64, 0, stream>>>(x, proj_w, proj_b, pbuf);
  k_pstats<<<MROWS / 256, 256, 0, stream>>>(pbuf, psum, psq, pminE, pmaxE);
  k_gatepre<<<1, 256, 0, stream>>>(gate_w, gate_b, G, wbv, colw, gbsc);
  k_spline<<<MROWS * NF / 256, 256, 0, stream>>>(pbuf, psum, psq, pminE, pmaxE,
                                                 pn_g, pn_b, knots, cps, tbuf);
  k_gbnstats<<<(SEQ + 255) / 256, 256, 0, stream>>>(tbuf, G, wbv, colw, gbsc, ssum, ssq);
  k_gluln<<<MROWS, 256, 0, stream>>>(tbuf, gate_w, gate_b, ssum, ssq,
                                     gbn_g, gbn_b, ln_g, ln_b, gatedh);

  // out = gated @ outp_w.T + outp_b (f32) + channel stats
  k_gemm<false><<<dim3(DMODEL / 128, MROWS / 128), 256, 0, stream>>>(
      gatedh, woutpT, outp_b, outbuf, nullptr, MROWS, DMODEL, DMODEL);
  k_cstats<<<MROWS / 128, 256, 0, stream>>>(outbuf, s2, q2);

  // residual combine with both BNs
  k_final<<<(MROWS * DMODEL) / 256, 256, 0, stream>>>(
      x, attn_o, outbuf, stats, attn_g, attn_b, on_g, on_b, out);
}
